// PvmNet_57750130262069
// MI455X (gfx1250) — compile-verified, matrix operand order checked by reading, not run on hardware
//
#include <hip/hip_runtime.h>
#include <cstdint>
#include <cstddef>

// ---------------------------------------------------------------- types
typedef _Float16 h16;
typedef __attribute__((ext_vector_type(16))) _Float16 v16h;
typedef __attribute__((ext_vector_type(8)))  _Float16 v8h;
typedef __attribute__((ext_vector_type(8)))  float    v8f;

#define GRES 64
#define NPTS 100000

static inline int cdivh(int a, int b) { return (a + b - 1) / b; }
static inline int cpad32(int c) { return ((c + 31) / 32) * 32; }

// ---------------------------------------------------------------- utility kernels
__global__ void k_zero(unsigned int* __restrict__ p, size_t n) {
  size_t i = (size_t)blockIdx.x * blockDim.x + threadIdx.x;
  if (i < n) p[i] = 0u;
}

// points_F (N,3) f32 -> (N,32) f16 zero padded
__global__ void k_cvt_points(const float* __restrict__ pf, h16* __restrict__ o, int np) {
  int i = blockIdx.x * blockDim.x + threadIdx.x;
  if (i >= np * 32) return;
  int p = i >> 5, c = i & 31;
  o[i] = (h16)(c < 3 ? pf[p * 3 + c] : 0.0f);
}

// point -> voxel scatter (sums + counts), f32 atomics
__global__ void k_voxelize(const float* __restrict__ coords, const h16* __restrict__ feats,
                           int np, int C, int Cp, float sinv, int g,
                           float* __restrict__ sums, float* __restrict__ cnt) {
  int p = blockIdx.x * blockDim.x + threadIdx.x;
  if (p >= np) return;
  int v0 = (int)floorf(coords[p * 3 + 0] * sinv);
  int v1 = (int)floorf(coords[p * 3 + 1] * sinv);
  int v2 = (int)floorf(coords[p * 3 + 2] * sinv);
  v0 = v0 < 0 ? 0 : (v0 > g - 1 ? g - 1 : v0);
  v1 = v1 < 0 ? 0 : (v1 > g - 1 ? g - 1 : v1);
  v2 = v2 < 0 ? 0 : (v2 > g - 1 ? g - 1 : v2);
  size_t flat = ((size_t)v0 * g + v1) * g + v2;
  atomicAdd(&cnt[flat], 1.0f);
  const h16* f = feats + (size_t)p * Cp;
  float* s = sums + flat * (size_t)Cp;
  for (int c = 0; c < C; ++c) atomicAdd(&s[c], (float)f[c]);
}

__global__ void k_vox_norm(const float* __restrict__ sums, const float* __restrict__ cnt,
                           h16* __restrict__ grid, int Cp, size_t total) {
  size_t i = (size_t)blockIdx.x * blockDim.x + threadIdx.x;
  if (i >= total) return;
  grid[i] = (h16)(sums[i] / fmaxf(cnt[i / (size_t)Cp], 1.0f));
}

// trilinear voxel -> point gather, optional fused add of a second point-feature array
__global__ void k_devox(const h16* __restrict__ grid, int g, int Cp, int C,
                        const float* __restrict__ coords, float sinv,
                        const h16* __restrict__ addf, h16* __restrict__ out, int np) {
  int p = blockIdx.x * blockDim.x + threadIdx.x;
  if (p >= np) return;
  float c0 = coords[p * 3 + 0] * sinv, c1 = coords[p * 3 + 1] * sinv, c2 = coords[p * 3 + 2] * sinv;
  float f0 = floorf(c0), f1 = floorf(c1), f2 = floorf(c2);
  float r0 = c0 - f0, r1 = c1 - f1, r2 = c2 - f2;
  int i0 = (int)f0, i1 = (int)f1, i2 = (int)f2;
  size_t cidx[8]; float wgt[8];
#pragma unroll
  for (int j = 0; j < 8; ++j) {
    int a0 = (j >> 2) & 1, a1 = (j >> 1) & 1, a2 = j & 1;
    int q0 = i0 + a0, q1 = i1 + a1, q2 = i2 + a2;
    q0 = q0 < 0 ? 0 : (q0 > g - 1 ? g - 1 : q0);
    q1 = q1 < 0 ? 0 : (q1 > g - 1 ? g - 1 : q1);
    q2 = q2 < 0 ? 0 : (q2 > g - 1 ? g - 1 : q2);
    cidx[j] = ((size_t)q0 * g + q1) * g + q2;
    wgt[j] = (a0 ? r0 : 1.0f - r0) * (a1 ? r1 : 1.0f - r1) * (a2 ? r2 : 1.0f - r2);
  }
  for (int cb = 0; cb < C; cb += 8) {
    float acc[8] = {0, 0, 0, 0, 0, 0, 0, 0};
#pragma unroll
    for (int j = 0; j < 8; ++j) {
      v8h v = *(const v8h*)(grid + cidx[j] * (size_t)Cp + cb);
#pragma unroll
      for (int e = 0; e < 8; ++e) acc[e] += wgt[j] * (float)v[e];
    }
    if (addf) {
      v8h av = *(const v8h*)(addf + (size_t)p * Cp + cb);
#pragma unroll
      for (int e = 0; e < 8; ++e) acc[e] += (float)av[e];
    }
    v8h o;
#pragma unroll
    for (int e = 0; e < 8; ++e) o[e] = (h16)acc[e];
    *(v8h*)(out + (size_t)p * Cp + cb) = o;
  }
}

// strided channel copy (for concat)
__global__ void k_copy_ch(const h16* __restrict__ src, int sCp, h16* __restrict__ dst,
                          int dCp, int dOff, int C, size_t nvox) {
  size_t i = (size_t)blockIdx.x * blockDim.x + threadIdx.x;
  size_t tot = nvox * (size_t)C;
  if (i >= tot) return;
  size_t v = i / (size_t)C;
  int c = (int)(i % (size_t)C);
  dst[v * dCp + dOff + c] = src[v * sCp + c];
}

__global__ void k_add_relu(const h16* __restrict__ a, const h16* __restrict__ b,
                           h16* __restrict__ o, size_t n) {
  size_t i = (size_t)blockIdx.x * blockDim.x + threadIdx.x;
  if (i >= n) return;
  float v = (float)a[i] + (float)b[i];
  o[i] = (h16)fmaxf(v, 0.0f);
}

// ---------------------------------------------------------------- weight packing
// Pack f32 weights (T taps, Cin, Cout) into per-wave WMMA B fragments (f16):
// fragment(nt, kb, t): 32 lanes x 16 halves; lane -> N = nt*16 + (lane&15),
// K group offset koff = (lane>=16)*8, half h -> K = kb*32 + koff + h + (h>=8 ? 8 : 0).
__global__ void k_pack_w(const float* __restrict__ W, h16* __restrict__ out,
                         int T, int Cin, int Cout, int KB, int NT) {
  size_t i = (size_t)blockIdx.x * blockDim.x + threadIdx.x;
  size_t tot = (size_t)NT * KB * T * 512;
  if (i >= tot) return;
  int h = (int)(i & 15);
  int lane = (int)((i >> 4) & 31);
  size_t rest = i >> 9;
  int t = (int)(rest % (size_t)T);
  int kb = (int)((rest / (size_t)T) % (size_t)KB);
  int nt = (int)(rest / ((size_t)T * KB));
  int koff = (lane >> 4) << 3;
  int k = kb * 32 + koff + h + (h >= 8 ? 8 : 0);
  int n = nt * 16 + (lane & 15);
  float v = (k < Cin && n < Cout) ? W[((size_t)t * Cin + k) * Cout + n] : 0.0f;
  out[i] = (h16)v;
}

// ---------------------------------------------------------------- implicit-GEMM conv / deconv / linear via WMMA
// One wave per 32(M) x 32(N) macro-tile = 2x2 WMMA tiles: each K-step loads 2 A + 2 B
// fragments and issues 4 independent v_wmma_f32_16x16x32_f16 (double the FLOP/byte of a
// 16x16 tile and 4-deep WMMA ILP to hide load latency).
// mode 0: conv (ks, stride, pad over m = output voxels)
// mode 1: deconv ks2 s2 (m = input voxels, tap = blockIdx.z, out voxel = 2*v + tap offset)
__global__ __launch_bounds__(32) void k_conv_wmma(
    const h16* __restrict__ in, const h16* __restrict__ wpk,
    const float* __restrict__ bn_g, const float* __restrict__ bn_b,
    const float* __restrict__ bn_m, const float* __restrict__ bn_v,
    const float* __restrict__ bias, const h16* __restrict__ resid,
    h16* __restrict__ out, float* __restrict__ out32,
    int Dm, int Hm, int Wm, int Din, int Hin, int Win,
    int CinPad, int KB, int Cout, int CoutPad,
    int ks, int stride, int pad, int relu, int mode) {
  const int lane = threadIdx.x;
  const int row = lane & 15;
  const int koff = (lane >> 4) << 3;
  const int mt = blockIdx.x, nt2 = blockIdx.y;
  const int Mtot = Dm * Hm * Wm;
  const int mbase = mt * 32;
  const int T = (mode == 0) ? ks * ks * ks : 1;
  const int Tpk = (mode == 0) ? T : 8;
  const int tap8 = (mode == 1) ? (int)blockIdx.z : 0;

  bool mval[2];
  int dd[2], hh[2], ww[2], mm0[2];
#pragma unroll
  for (int s = 0; s < 2; ++s) {
    int m = mbase + s * 16 + row;
    mval[s] = (m < Mtot);
    mm0[s] = mval[s] ? m : 0;
    dd[s] = mm0[s] / (Hm * Wm);
    hh[s] = (mm0[s] / Wm) % Hm;
    ww[s] = mm0[s] % Wm;
  }

  v8f acc[2][2];
#pragma unroll
  for (int s = 0; s < 2; ++s)
#pragma unroll
    for (int u = 0; u < 2; ++u) acc[s][u] = (v8f){0.f, 0.f, 0.f, 0.f, 0.f, 0.f, 0.f, 0.f};
  const v16h vzero = {};

  for (int t = 0; t < T; ++t) {
    bool inb[2];
    const h16* rp[2];
#pragma unroll
    for (int s = 0; s < 2; ++s) {
      size_t inIdx;
      if (mode == 0) {
        int td = t / (ks * ks), th = (t / ks) % ks, tw = t % ks;
        int id = dd[s] * stride - pad + td;
        int ih = hh[s] * stride - pad + th;
        int iw = ww[s] * stride - pad + tw;
        inb[s] = mval[s] && (unsigned)id < (unsigned)Din && (unsigned)ih < (unsigned)Hin &&
                 (unsigned)iw < (unsigned)Win;
        inIdx = inb[s] ? ((size_t)((size_t)id * Hin + ih) * Win + iw) : (size_t)0;
      } else {
        inb[s] = mval[s];
        inIdx = (size_t)mm0[s];
      }
      rp[s] = in + inIdx * (size_t)CinPad;
    }
    const int tsel = (mode == 0) ? t : tap8;
#pragma unroll 2
    for (int kb = 0; kb < KB; ++kb) {
      v16h A[2];
#pragma unroll
      for (int s = 0; s < 2; ++s) {
        const h16* ap = rp[s] + kb * 32 + koff;
        v8h alo = *(const v8h*)(ap);
        v8h ahi = *(const v8h*)(ap + 16);
#pragma unroll
        for (int e = 0; e < 8; ++e) { A[s][e] = alo[e]; A[s][8 + e] = ahi[e]; }
        if (!inb[s]) A[s] = vzero;
      }
      v16h B[2];
#pragma unroll
      for (int u = 0; u < 2; ++u) {
        const h16* bp =
            wpk + (((size_t)(nt2 * 2 + u) * KB + kb) * Tpk + tsel) * 512 + lane * 16;
        __builtin_prefetch(bp + (size_t)Tpk * 512, 0, 1);  // next kb's fragment
        B[u] = *(const v16h*)bp;
      }
      acc[0][0] = __builtin_amdgcn_wmma_f32_16x16x32_f16(false, A[0], false, B[0], (short)0,
                                                         acc[0][0], false, false);
      acc[0][1] = __builtin_amdgcn_wmma_f32_16x16x32_f16(false, A[0], false, B[1], (short)0,
                                                         acc[0][1], false, false);
      acc[1][0] = __builtin_amdgcn_wmma_f32_16x16x32_f16(false, A[1], false, B[0], (short)0,
                                                         acc[1][0], false, false);
      acc[1][1] = __builtin_amdgcn_wmma_f32_16x16x32_f16(false, A[1], false, B[1], (short)0,
                                                         acc[1][1], false, false);
    }
  }

  // ------------- epilogue: BN (or bias), optional residual add, ReLU, store
  const int ncol = lane & 15;
  const int rbase = (lane >> 4) << 3;  // lanes 16..31 hold rows M=8..15
#pragma unroll
  for (int u = 0; u < 2; ++u) {
    int nglob = nt2 * 32 + u * 16 + ncol;
    bool nok = nglob < Cout;
    int ng = nok ? nglob : 0;
    float sc = 1.0f, bi = 0.0f;
    if (bn_g) {
      float s = bn_g[ng] * rsqrtf(bn_v[ng] + 1e-5f);
      sc = s;
      bi = bn_b[ng] - bn_m[ng] * s;
    } else if (bias) {
      bi = bias[ng];
    }
#pragma unroll
    for (int s = 0; s < 2; ++s) {
#pragma unroll
      for (int r = 0; r < 8; ++r) {
        int mr = mbase + s * 16 + rbase + r;
        if (mr < Mtot && nok) {
          size_t ov;
          if (mode == 0) {
            ov = (size_t)mr;
          } else {
            int d2 = mr / (Hm * Wm), h2 = (mr / Wm) % Hm, w2 = mr % Wm;
            int td = tap8 >> 2, th = (tap8 >> 1) & 1, tw = tap8 & 1;
            ov = ((size_t)(2 * d2 + td) * (2 * Hm) + (2 * h2 + th)) * (size_t)(2 * Wm) +
                 (2 * w2 + tw);
          }
          float v = acc[s][u][r] * sc + bi;
          if (resid) v += (float)resid[ov * (size_t)CoutPad + nglob];
          if (relu) v = fmaxf(v, 0.0f);
          if (out32) out32[ov * (size_t)Cout + nglob] = v;
          else out[ov * (size_t)CoutPad + nglob] = (h16)v;
        }
      }
    }
  }
}

// ---------------------------------------------------------------- host orchestration
extern "C" void kernel_launch(void* const* d_in, const int* in_sizes, int n_in,
                              void* d_out, int out_size, void* d_ws, size_t ws_size,
                              hipStream_t stream) {
  (void)in_sizes; (void)n_in; (void)out_size; (void)ws_size;

  // ---- parse inputs in setup_inputs() dict order ----
  int ip = 0;
  auto F = [&]() { return (const float*)d_in[ip++]; };
  const float* points_F = F();
  const float* points_C = F();
  struct BN { const float *g, *b, *m, *v; };
  auto getBN = [&]() { BN x; x.g = F(); x.b = F(); x.m = F(); x.v = F(); return x; };
  struct ConvP { const float* w; BN bn; };
  auto getCW = [&]() { ConvP c; c.w = F(); c.bn = getBN(); return c; };
  ConvP in1 = getCW(), in2 = getCW();
  struct Res { ConvP main; const float* dw; BN dbn; bool hasD; };
  struct StageP { ConvP c; Res r1; Res r2; };
  StageP st[6];
  for (int i = 0; i < 6; ++i) {
    st[i].c = getCW();
    st[i].r1.main = getCW(); st[i].r1.dw = F(); st[i].r1.dbn = getBN(); st[i].r1.hasD = true;
    st[i].r2.main = getCW(); st[i].r2.dw = nullptr; st[i].r2.hasD = false;
  }
  struct UpP { ConvP d; Res r1; Res r2; };
  UpP up[6];
  for (int i = 0; i < 6; ++i) {
    up[i].d = getCW();
    up[i].r1.main = getCW(); up[i].r1.dw = F(); up[i].r1.dbn = getBN(); up[i].r1.hasD = true;
    up[i].r2.main = getCW(); up[i].r2.dw = nullptr; up[i].r2.hasD = false;
  }
  ConvP pt[4];
  for (int i = 0; i < 4; ++i) pt[i] = getCW();
  const float* outW = F();
  const float* outB = F();

  // ---- workspace arena ----
  char* base = (char*)d_ws;
  size_t cur = 0;
  auto alloc = [&](size_t bytes) -> void* {
    size_t a = (cur + 255) & ~(size_t)255;
    void* p = base + a;
    cur = a + bytes;
    return p;
  };
  auto zero = [&](void* p, size_t bytes) {
    size_t n = bytes / 4;
    k_zero<<<(unsigned)((n + 255) / 256), 256, 0, stream>>>((unsigned int*)p, n);
  };

  // ---- launch helpers ----
  auto pack = [&](const float* W, int T, int Cin, int Cout) -> h16* {
    int KB = cdivh(Cin, 32), NT = cdivh(Cout, 16);
    size_t tot = (size_t)NT * KB * T * 512;
    h16* p = (h16*)alloc(tot * 2);
    k_pack_w<<<(unsigned)((tot + 255) / 256), 256, 0, stream>>>(W, p, T, Cin, Cout, KB, NT);
    return p;
  };
  auto conv = [&](const h16* in, const float* W, const BN* bn, const float* bias,
                  const h16* resid, h16* out, float* out32,
                  int Dm, int Hm, int Wm, int Din, int Hin, int Win,
                  int Cin, int Cout, int ks, int stride, int pad, int relu, int mode,
                  int coutPadOv = 0) {
    int KB = cdivh(Cin, 32), NT = cdivh(Cout, 16);  // NT is even for every layer here
    int T = mode ? 8 : ks * ks * ks;
    h16* wpk = pack(W, T, Cin, Cout);
    int Mtot = Dm * Hm * Wm;
    dim3 grid((unsigned)cdivh(Mtot, 32), (unsigned)(NT / 2), mode ? 8u : 1u);
    int CoutPad = coutPadOv ? coutPadOv : cpad32(Cout);
    k_conv_wmma<<<grid, 32, 0, stream>>>(
        in, wpk, bn ? bn->g : nullptr, bn ? bn->b : nullptr, bn ? bn->m : nullptr,
        bn ? bn->v : nullptr, bias, resid, out, out32, Dm, Hm, Wm, Din, Hin, Win,
        cpad32(Cin), KB, Cout, CoutPad, ks, stride, pad, relu, mode);
  };
  auto resblk = [&](const h16* x, const Res& r, h16* tmpH, h16* out, int S, int Cin,
                    int Cout) {
    conv(x, r.main.w, &r.main.bn, nullptr, nullptr, tmpH, nullptr, S, S, S, S, S, S, Cin,
         Cout, 3, 1, 1, 1, 0);
    if (r.hasD) {
      conv(x, r.dw, &r.dbn, nullptr, tmpH, out, nullptr, S, S, S, S, S, S, Cin, Cout, 1, 1,
           0, 1, 0);
    } else {
      size_t n = (size_t)S * S * S * (size_t)cpad32(Cout);
      k_add_relu<<<(unsigned)((n + 255) / 256), 256, 0, stream>>>(tmpH, x, out, n);
    }
  };
  float* voxsum = (float*)alloc((size_t)GRES * GRES * GRES * 32 * 4);
  float* voxcnt = (float*)alloc((size_t)GRES * GRES * GRES * 4);
  auto ptv = [&](const h16* feats, int C, int g, float strideval, h16* grid) {
    int Cp = cpad32(C);
    size_t vox = (size_t)g * g * g;
    zero(voxsum, vox * Cp * 4);
    zero(voxcnt, vox * 4);
    k_voxelize<<<(unsigned)cdivh(NPTS, 256), 256, 0, stream>>>(
        points_C, feats, NPTS, C, Cp, 1.0f / strideval, g, voxsum, voxcnt);
    size_t tot = vox * (size_t)Cp;
    k_vox_norm<<<(unsigned)((tot + 255) / 256), 256, 0, stream>>>(voxsum, voxcnt, grid, Cp,
                                                                  tot);
  };
  auto v2p = [&](const h16* grid, int g, int C, float strideval, const h16* addf, h16* out) {
    k_devox<<<(unsigned)cdivh(NPTS, 256), 256, 0, stream>>>(
        grid, g, cpad32(C), C, points_C, 1.0f / strideval, addf, out, NPTS);
  };
  auto ptrans = [&](const h16* z, const ConvP& p, int Cin, int Cout, h16* out) {
    conv(z, p.w, &p.bn, nullptr, nullptr, out, nullptr, 1, 1, NPTS, 1, 1, NPTS, Cin, Cout,
         1, 1, 0, 1, 0);
  };

  // ---- persistent activation buffers ----
  const size_t gvox = (size_t)GRES * GRES * GRES;
  h16* pfeat = (h16*)alloc((size_t)NPTS * 32 * 2);
  h16* gridIn = (h16*)alloc(gvox * 32 * 2);
  h16* x0a = (h16*)alloc(gvox * 32 * 2);
  h16* x0 = (h16*)alloc(gvox * 32 * 2);
  h16* xg = (h16*)alloc(gvox * 32 * 2);

  const int DOWNci[6] = {32, 64, 96, 128, 160, 192};
  const int DOWNco[6] = {64, 96, 128, 160, 192, 224};
  h16 *stg_c[6], *stg_h[6], *stg_r1[6], *stg_h2[6], *xs[6];
  {
    int S = GRES;
    for (int i = 0; i < 6; ++i) {
      int So = S / 2;
      size_t v = (size_t)So * So * So;
      stg_c[i] = (h16*)alloc(v * DOWNci[i] * 2);
      stg_h[i] = (h16*)alloc(v * DOWNco[i] * 2);
      stg_r1[i] = (h16*)alloc(v * DOWNco[i] * 2);
      stg_h2[i] = (h16*)alloc(v * DOWNco[i] * 2);
      xs[i] = (h16*)alloc(v * DOWNco[i] * 2);
      S = So;
    }
  }
  h16* z0 = (h16*)alloc((size_t)NPTS * 32 * 2);
  h16* z1 = (h16*)alloc((size_t)NPTS * 224 * 2);
  h16* z2 = (h16*)alloc((size_t)NPTS * 160 * 2);
  h16* z3 = (h16*)alloc((size_t)NPTS * 96 * 2);
  h16* z4 = (h16*)alloc((size_t)NPTS * 32 * 2);
  h16* ptb = (h16*)alloc((size_t)NPTS * 224 * 2);

  h16* yg0 = (h16*)alloc((size_t)1 * 224 * 2);     // ptv(z1, 64, g=1)
  h16* yg1 = (h16*)alloc((size_t)64 * 160 * 2);    // ptv(z2, 16, g=4)
  h16* yg2 = (h16*)alloc((size_t)4096 * 96 * 2);   // ptv(z3, 4, g=16)

  const int UPco[6] = {192, 160, 128, 96, 64, 32};
  const int upSin[6] = {1, 2, 4, 8, 16, 32};
  h16 *ucat[6], *uh[6], *ur1[6], *uh2[6], *uout[6];
  for (int i = 0; i < 6; ++i) {
    int So = 2 * upSin[i];
    size_t v = (size_t)So * So * So;
    ucat[i] = (h16*)alloc(v * 2 * UPco[i] * 2);
    uh[i] = (h16*)alloc(v * UPco[i] * 2);
    ur1[i] = (h16*)alloc(v * UPco[i] * 2);
    uh2[i] = (h16*)alloc(v * UPco[i] * 2);
    uout[i] = (h16*)alloc(v * UPco[i] * 2);
  }
  auto upblk = [&](const h16* y, int Sin, int Cin, int ui, int Cout, const h16* skip) {
    int So = 2 * Sin;
    size_t vox = (size_t)So * So * So;
    // deconv (+BN+ReLU) writes channels [0,Cout) of the concat buffer (stride 2*Cout)
    conv(y, up[ui].d.w, &up[ui].d.bn, nullptr, nullptr, ucat[ui], nullptr, Sin, Sin, Sin,
         Sin, Sin, Sin, Cin, Cout, 2, 2, 0, 1, 1, 2 * Cout);
    size_t tot = vox * (size_t)Cout;
    k_copy_ch<<<(unsigned)((tot + 255) / 256), 256, 0, stream>>>(
        skip, cpad32(Cout), ucat[ui], 2 * Cout, Cout, Cout, vox);
    resblk(ucat[ui], up[ui].r1, uh[ui], ur1[ui], So, 2 * Cout, Cout);
    resblk(ur1[ui], up[ui].r2, uh2[ui], uout[ui], So, Cout, Cout);
  };

  // ================= pipeline =================
  k_cvt_points<<<(unsigned)cdivh(NPTS * 32, 256), 256, 0, stream>>>(points_F, pfeat, NPTS);
  ptv(pfeat, 3, GRES, 1.0f, gridIn);
  conv(gridIn, in1.w, &in1.bn, nullptr, nullptr, x0a, nullptr, GRES, GRES, GRES, GRES, GRES,
       GRES, 3, 32, 3, 1, 1, 1, 0);
  conv(x0a, in2.w, &in2.bn, nullptr, nullptr, x0, nullptr, GRES, GRES, GRES, GRES, GRES,
       GRES, 32, 32, 3, 1, 1, 1, 0);
  v2p(x0, GRES, 32, 1.0f, nullptr, z0);
  ptv(z0, 32, GRES, 1.0f, xg);

  const h16* cx = xg;
  int S = GRES;
  for (int i = 0; i < 6; ++i) {
    int ci = DOWNci[i], co = DOWNco[i];
    int So = S / 2;
    conv(cx, st[i].c.w, &st[i].c.bn, nullptr, nullptr, stg_c[i], nullptr, So, So, So, S, S,
         S, ci, ci, 2, 2, 0, 1, 0);
    resblk(stg_c[i], st[i].r1, stg_h[i], stg_r1[i], So, ci, co);
    resblk(stg_r1[i], st[i].r2, stg_h2[i], xs[i], So, co, co);
    cx = xs[i];
    S = So;
  }

  ptrans(z0, pt[0], 32, 224, ptb);
  v2p(xs[5], 1, 224, 64.0f, ptb, z1);
  ptv(z1, 224, 1, 64.0f, yg0);
  upblk(yg0, 1, 224, 0, 192, xs[4]);
  upblk(uout[0], 2, 192, 1, 160, xs[3]);

  ptrans(z1, pt[1], 224, 160, ptb);
  v2p(uout[1], 4, 160, 16.0f, ptb, z2);
  ptv(z2, 160, 4, 16.0f, yg1);
  upblk(yg1, 4, 160, 2, 128, xs[2]);
  upblk(uout[2], 8, 128, 3, 96, xs[1]);

  ptrans(z2, pt[2], 160, 96, ptb);
  v2p(uout[3], 16, 96, 4.0f, ptb, z3);
  ptv(z3, 96, 16, 4.0f, yg2);
  upblk(yg2, 16, 96, 4, 64, xs[0]);
  upblk(uout[4], 32, 64, 5, 32, x0);

  ptrans(z3, pt[3], 96, 32, ptb);
  v2p(uout[5], GRES, 32, 1.0f, ptb, z4);

  // final classifier: z4 @ W + b -> f32 d_out (100000 x 20)
  conv(z4, outW, nullptr, outB, nullptr, nullptr, (float*)d_out, 1, 1, NPTS, 1, 1, NPTS, 32,
       20, 1, 1, 0, 0, 0);
}
